// AnomalyAttention_10411000725522
// MI455X (gfx1250) — matrix operand order repaired and checked by reading, compiled, not run on hardware
//
#include <hip/hip_runtime.h>
#include <math.h>

// Problem constants (from reference): B=16, N=1024, C_IN=C_OUT=256
#define B_   16
#define N_   1024
#define D_   256
#define BN_  (B_ * N_)          // 16384 rows of x
#define NN_  (N_ * N_)          // 1048576

typedef __attribute__((ext_vector_type(16))) __bf16 v16bf;
typedef __attribute__((ext_vector_type(8)))  __bf16 v8bf;
typedef __attribute__((ext_vector_type(8)))  float  v8f;

#define WMMA_BF16(a, b, c) \
  __builtin_amdgcn_wmma_f32_16x16x32_bf16(false, (a), false, (b), (short)0, (c), false, false)

// ---------------------------------------------------------------------------
// Async global -> LDS copy (GLOBAL_LOAD_ASYNC_TO_LDS_B128, ASYNCcnt-tracked).
// Builtin signature (from hipcc diagnostic): pointee type is int __vector(4),
// args = (global src, lds dst, imm offset, imm cpol).
// ---------------------------------------------------------------------------
#if defined(__has_builtin)
#if __has_builtin(__builtin_amdgcn_global_load_async_to_lds_b128)
#define HAVE_ASYNC_LDS 1
#endif
#endif
#ifndef HAVE_ASYNC_LDS
#define HAVE_ASYNC_LDS 0
#endif

typedef int v4i_ __attribute__((vector_size(16)));
typedef __attribute__((address_space(1))) v4i_ as1_v4i;
typedef __attribute__((address_space(3))) v4i_ as3_v4i;

__device__ __forceinline__ void cp16_g2l(void* lds_dst, const void* g_src) {
#if HAVE_ASYNC_LDS
  __builtin_amdgcn_global_load_async_to_lds_b128((as1_v4i*)g_src, (as3_v4i*)lds_dst,
                                                 /*imm offset*/ 0, /*cpol*/ 0);
#else
  *(int4*)lds_dst = *(const int4*)g_src;
#endif
}

__device__ __forceinline__ void async_wait0() {
#if HAVE_ASYNC_LDS
#if __has_builtin(__builtin_amdgcn_s_wait_asynccnt)
  __builtin_amdgcn_s_wait_asynccnt(0);
#else
  asm volatile("s_wait_asynccnt 0x0" ::: "memory");
#endif
#endif
}

// ---------------------------------------------------------------------------
// Tile staging: 128 x 32 tiles, K-contiguous rows (row stride in LDS = 32 elems)
// bf16 tile = 8 KB = 512 x 16B chunks; fp32 tile = 16 KB = 1024 x 16B chunks.
// 256 threads per block.
// ---------------------------------------------------------------------------
__device__ __forceinline__ void stage_bf16_128x32(__bf16* lds, const __bf16* g, int ld) {
#pragma unroll
  for (int c = threadIdx.x; c < 512; c += 256) {
    const int row = c >> 2;             // 4 chunks (64B) per row
    const int off = (c & 3) * 8;        // bf16 elems
    cp16_g2l(lds + row * 32 + off, g + (size_t)row * ld + off);
  }
}

__device__ __forceinline__ void stage_f32_128x32(float* lds, const float* g, int ld) {
#pragma unroll
  for (int c = threadIdx.x; c < 1024; c += 256) {
    const int row = c >> 3;             // 8 chunks (128B) per row
    const int off = (c & 7) * 4;        // float elems
    cp16_g2l(lds + row * 32 + off, g + (size_t)row * ld + off);
  }
}

// ---------------------------------------------------------------------------
// WMMA fragment builders from LDS (wave32, layouts per CDNA5 ISA)
// A (16x32 MxK): lanes 0-15 M=lane, K elems {0..7,16..23}; lanes 16-31 +8.
// B (32x16 KxN): lane n = N; lanes 0-15 K elems {0..7,16..23}; lanes 16-31 +8.
// LDS tiles are [row][32] with K contiguous.
// ---------------------------------------------------------------------------
__device__ __forceinline__ v16bf frag_a_lds_bf16(const __bf16* lds, int row0) {
  const int lane = threadIdx.x & 31;
  const int m = lane & 15;
  const int hsel = lane >> 4;
  const __bf16* p = lds + (row0 + m) * 32 + hsel * 8;
  v8bf lo = *(const v8bf*)(p);
  v8bf hi = *(const v8bf*)(p + 16);
  v16bf r;
#pragma unroll
  for (int i = 0; i < 8; ++i) { r[i] = lo[i]; r[i + 8] = hi[i]; }
  return r;
}

__device__ __forceinline__ v16bf frag_a_lds_f32(const float* lds, int row0) {
  const int lane = threadIdx.x & 31;
  const int m = lane & 15;
  const int hsel = lane >> 4;
  const float* p = lds + (row0 + m) * 32 + hsel * 8;
  v8f lo = *(const v8f*)(p);
  v8f hi = *(const v8f*)(p + 16);
  v16bf r;
#pragma unroll
  for (int i = 0; i < 8; ++i) { r[i] = (__bf16)lo[i]; r[i + 8] = (__bf16)hi[i]; }
  return r;
}

__device__ __forceinline__ v16bf frag_b_lds(const __bf16* lds, int col0) {
  const int lane = threadIdx.x & 31;
  const int n = lane & 15;
  const int ks = (lane >> 4) * 8;
  const __bf16* p = lds + (col0 + n) * 32 + ks;
  v8bf lo = *(const v8bf*)(p);
  v8bf hi = *(const v8bf*)(p + 16);
  v16bf r;
#pragma unroll
  for (int i = 0; i < 8; ++i) { r[i] = lo[i]; r[i + 8] = hi[i]; }
  return r;
}

// ---------------------------------------------------------------------------
// Stage 0: conversions
// ---------------------------------------------------------------------------
__global__ void cvt_bf16_kernel(const float* __restrict__ in, __bf16* __restrict__ out, int n) {
  int i = blockIdx.x * blockDim.x + threadIdx.x;
  const int stride = gridDim.x * blockDim.x;
  for (; i < n; i += stride) out[i] = (__bf16)in[i];
}

// Wt[n][k] = W[k][n]  (256x256), bf16, so GEMM B-operand is K-contiguous
__global__ void transpose_w_kernel(const float* __restrict__ W, __bf16* __restrict__ Wt) {
  const int idx = blockIdx.x * blockDim.x + threadIdx.x;   // 65536
  const int n = idx >> 8, k = idx & 255;
  Wt[idx] = (__bf16)W[k * 256 + n];
}

// sigma[row] = dot(x[row,:], W_sigma), one wave per row
__global__ __launch_bounds__(256) void sigma_kernel(const float* __restrict__ x,
                                                    const float* __restrict__ Wsig,
                                                    float* __restrict__ sig) {
  const int row  = blockIdx.x * 8 + (threadIdx.x >> 5);
  const int lane = threadIdx.x & 31;
  const float* xr = x + (size_t)row * D_;
  float s = 0.f;
#pragma unroll
  for (int t = 0; t < 8; ++t) s += xr[lane + t * 32] * Wsig[lane + t * 32];
#pragma unroll
  for (int off = 16; off > 0; off >>= 1) s += __shfl_down(s, off, 32);
  if (lane == 0) sig[row] = s;
}

// ---------------------------------------------------------------------------
// Stage 1: QKV projection.  C[16384,256] = Xb @ W (B-operand = Wt, K-contig).
// 128x128 block tile, double-buffered async LDS staging, 8 waves each owning
// a 16x128 strip (8 accumulators -> each A frag feeds 8 WMMAs).
// transpose_out=1 stores V as Vt[b][d][m].
// ---------------------------------------------------------------------------
__global__ __launch_bounds__(256) void qkv_gemm_kernel(const __bf16* __restrict__ X,
                                                       const __bf16* __restrict__ Wt,
                                                       __bf16* __restrict__ Out,
                                                       int transpose_out) {
  __shared__ alignas(32) __bf16 As[2][128 * 32];
  __shared__ alignas(32) __bf16 Bs[2][128 * 32];
  const int cb = blockIdx.x & 1;          // N=256 -> 2 col blocks
  const int rb = blockIdx.x >> 1;         // 128 row blocks
  const int row_base = rb * 128, col_base = cb * 128;
  const int wave = threadIdx.x >> 5;

  v8f acc[8] = {};
  stage_bf16_128x32(As[0], X  + (size_t)row_base * D_, D_);
  stage_bf16_128x32(Bs[0], Wt + (size_t)col_base * D_, D_);

  for (int kk = 0; kk < 8; ++kk) {        // K = 256 in steps of 32
    const int cur = kk & 1;
    async_wait0();
    __syncthreads();
    if (kk < 7) {
      const int knext = (kk + 1) * 32;
      stage_bf16_128x32(As[cur ^ 1], X  + (size_t)row_base * D_ + knext, D_);
      stage_bf16_128x32(Bs[cur ^ 1], Wt + (size_t)col_base * D_ + knext, D_);
    }
    v16bf a = frag_a_lds_bf16(As[cur], wave * 16);
#pragma unroll
    for (int j = 0; j < 8; ++j) {
      v16bf b = frag_b_lds(Bs[cur], j * 16);
      acc[j] = WMMA_BF16(a, b, acc[j]);
    }
    __syncthreads();
  }

  // C/D layout: VGPR r -> M = r + (lane>=16 ? 8 : 0), N = lane&15
  const int lane = threadIdx.x & 31;
  const int nl = lane & 15;
  const int mo = (lane >> 4) * 8;
  if (!transpose_out) {
#pragma unroll
    for (int j = 0; j < 8; ++j)
#pragma unroll
      for (int r = 0; r < 8; ++r)
        Out[(size_t)(row_base + wave * 16 + r + mo) * D_ + col_base + j * 16 + nl] =
            (__bf16)acc[j][r];
  } else {
#pragma unroll
    for (int j = 0; j < 8; ++j)
#pragma unroll
      for (int r = 0; r < 8; ++r) {
        const int row = row_base + wave * 16 + r + mo;   // b*1024 + m
        const int b = row >> 10, mloc = row & 1023;
        Out[(size_t)b * (D_ * N_) + (size_t)(col_base + j * 16 + nl) * N_ + mloc] =
            (__bf16)acc[j][r];
      }
  }
}

// ---------------------------------------------------------------------------
// Stage 2: scores[b] = Q[b] @ K[b]^T / 16, written into S region (pre-softmax).
// Same 128x128 block structure; B-operand rows are K[b][m][:] (K-contiguous).
// ---------------------------------------------------------------------------
__global__ __launch_bounds__(256) void scores_kernel(const __bf16* __restrict__ Q,
                                                     const __bf16* __restrict__ K,
                                                     float* __restrict__ S) {
  __shared__ alignas(32) __bf16 As[2][128 * 32];
  __shared__ alignas(32) __bf16 Bs[2][128 * 32];
  const int b  = blockIdx.x >> 6;         // 64 tiles per batch
  const int t  = blockIdx.x & 63;
  const int rb = t >> 3, cb = t & 7;      // 8x8 grid of 128x128 tiles
  const int row_base = rb * 128, col_base = cb * 128;
  const int wave = threadIdx.x >> 5;
  const __bf16* Qb = Q + (size_t)b * N_ * D_;
  const __bf16* Kb = K + (size_t)b * N_ * D_;

  v8f acc[8] = {};
  stage_bf16_128x32(As[0], Qb + (size_t)row_base * D_, D_);
  stage_bf16_128x32(Bs[0], Kb + (size_t)col_base * D_, D_);

  for (int kk = 0; kk < 8; ++kk) {
    const int cur = kk & 1;
    async_wait0();
    __syncthreads();
    if (kk < 7) {
      const int knext = (kk + 1) * 32;
      stage_bf16_128x32(As[cur ^ 1], Qb + (size_t)row_base * D_ + knext, D_);
      stage_bf16_128x32(Bs[cur ^ 1], Kb + (size_t)col_base * D_ + knext, D_);
    }
    v16bf a = frag_a_lds_bf16(As[cur], wave * 16);
#pragma unroll
    for (int j = 0; j < 8; ++j) {
      v16bf bfrag = frag_b_lds(Bs[cur], j * 16);
      acc[j] = WMMA_BF16(a, bfrag, acc[j]);
    }
    __syncthreads();
  }

  const float scale = 0.0625f;            // 1/sqrt(256)
  const int lane = threadIdx.x & 31;
  const int nl = lane & 15;
  const int mo = (lane >> 4) * 8;
  float* Sb = S + (size_t)b * NN_;
#pragma unroll
  for (int j = 0; j < 8; ++j)
#pragma unroll
    for (int r = 0; r < 8; ++r)
      Sb[(size_t)(row_base + wave * 16 + r + mo) * N_ + col_base + j * 16 + nl] =
          acc[j][r] * scale;
}

// ---------------------------------------------------------------------------
// Stage 3: softmax over the BATCH axis (16 values, stride N*N), in place.
// ---------------------------------------------------------------------------
__global__ __launch_bounds__(256) void softmax_batch_kernel(float* __restrict__ S) {
  const size_t idx = (size_t)blockIdx.x * blockDim.x + threadIdx.x;  // over N*N
  float v[B_];
  float mx = -3.0e38f;
#pragma unroll
  for (int b = 0; b < B_; ++b) { v[b] = S[(size_t)b * NN_ + idx]; mx = fmaxf(mx, v[b]); }
  float s = 0.f;
#pragma unroll
  for (int b = 0; b < B_; ++b) { v[b] = __expf(v[b] - mx); s += v[b]; }
  const float inv = 1.0f / s;
#pragma unroll
  for (int b = 0; b < B_; ++b) S[(size_t)b * NN_ + idx] = v[b] * inv;
}

// ---------------------------------------------------------------------------
// Stage 4: Z[b] = S[b] @ V[b].  A = S fp32 staged raw to LDS (async copy is
// byte-exact), converted to bf16 while building fragments.  B = Vt[b][d][m]
// so K(=m) is contiguous.  K = 1024 in 32 steps.
// ---------------------------------------------------------------------------
__global__ __launch_bounds__(256) void z_gemm_kernel(const float* __restrict__ S,
                                                     const __bf16* __restrict__ Vt,
                                                     float* __restrict__ Z) {
  __shared__ alignas(32) float  Asf[2][128 * 32];   // 32 KB
  __shared__ alignas(32) __bf16 Bs[2][128 * 32];    // 16 KB
  const int b  = blockIdx.x >> 4;         // 16 tiles per batch (8 row x 2 col)
  const int t  = blockIdx.x & 15;
  const int rb = t >> 1, cb = t & 1;
  const int row_base = rb * 128, col_base = cb * 128;
  const int wave = threadIdx.x >> 5;
  const float*  Sb = S  + (size_t)b * NN_;
  const __bf16* Vb = Vt + (size_t)b * D_ * N_;

  v8f acc[8] = {};
  stage_f32_128x32(Asf[0], Sb + (size_t)row_base * N_, N_);
  stage_bf16_128x32(Bs[0], Vb + (size_t)col_base * N_, N_);

  for (int kk = 0; kk < 32; ++kk) {       // K = 1024 in steps of 32
    const int cur = kk & 1;
    async_wait0();
    __syncthreads();
    if (kk < 31) {
      const int knext = (kk + 1) * 32;
      stage_f32_128x32(Asf[cur ^ 1], Sb + (size_t)row_base * N_ + knext, N_);
      stage_bf16_128x32(Bs[cur ^ 1], Vb + (size_t)col_base * N_ + knext, N_);
    }
    v16bf a = frag_a_lds_f32(Asf[cur], wave * 16);
#pragma unroll
    for (int j = 0; j < 8; ++j) {
      v16bf bfrag = frag_b_lds(Bs[cur], j * 16);
      acc[j] = WMMA_BF16(a, bfrag, acc[j]);
    }
    __syncthreads();
  }

  float* Zb = Z + (size_t)b * N_ * D_;
  const int lane = threadIdx.x & 31;
  const int nl = lane & 15;
  const int mo = (lane >> 4) * 8;
#pragma unroll
  for (int j = 0; j < 8; ++j)
#pragma unroll
    for (int r = 0; r < 8; ++r)
      Zb[(size_t)(row_base + wave * 16 + r + mo) * D_ + col_base + j * 16 + nl] = acc[j][r];
}

// ---------------------------------------------------------------------------
// Stage 5: prior association P.  One 256-thread block per (b,i) row.
// ---------------------------------------------------------------------------
__global__ __launch_bounds__(256) void prior_kernel(const float* __restrict__ eps,
                                                    const float* __restrict__ sigma,
                                                    float* __restrict__ P) {
  __shared__ float red[256];
  const int bi = blockIdx.x;              // b*1024 + i
  const int i  = bi & 1023;
  const float sg = fabsf(sigma[bi]);
  const float* e = eps + (size_t)bi * N_;
  float g[4];
  float psum = 0.f;
#pragma unroll
  for (int r = 0; r < 4; ++r) {
    const int j = threadIdx.x + r * 256;
    g[r] = fabsf((float)(i - j)) + sg * e[j];
    psum += g[r];
  }
  red[threadIdx.x] = psum;
  __syncthreads();
  for (int off = 128; off > 0; off >>= 1) {
    if ((int)threadIdx.x < off) red[threadIdx.x] += red[threadIdx.x + off];
    __syncthreads();
  }
  const float inv = 1.0f / red[0];
  float* Pr = P + (size_t)bi * N_;
#pragma unroll
  for (int r = 0; r < 4; ++r)
    Pr[threadIdx.x + r * 256] = g[r] * inv;
}

// ---------------------------------------------------------------------------
// Host launcher
// ---------------------------------------------------------------------------
extern "C" void kernel_launch(void* const* d_in, const int* in_sizes, int n_in,
                              void* d_out, int out_size, void* d_ws, size_t ws_size,
                              hipStream_t stream) {
  const float* x      = (const float*)d_in[0];   // [16,1024,256]
  const float* Wq     = (const float*)d_in[1];   // [256,256]
  const float* Wk     = (const float*)d_in[2];
  const float* Wv     = (const float*)d_in[3];
  const float* Wsigma = (const float*)d_in[4];   // [256,1]
  const float* eps    = (const float*)d_in[5];   // [16,1024,1024]

  // Outputs concatenated: Z [16,1024,256], P [16,1024,1024], S [16,1024,1024]
  float* Z = (float*)d_out;
  float* P = Z + (size_t)B_ * N_ * D_;
  float* S = P + (size_t)B_ * N_ * N_;

  // Workspace layout (~34 MB)
  char* ws = (char*)d_ws;
  __bf16* XB  = (__bf16*)(ws);                     //  8 MB  x bf16 [16384][256]
  __bf16* QB  = (__bf16*)(ws + 8388608u);          //  8 MB  Q bf16
  __bf16* KB  = (__bf16*)(ws + 16777216u);         //  8 MB  K bf16
  __bf16* VT  = (__bf16*)(ws + 25165824u);         //  8 MB  V^T bf16 [b][d][m]
  __bf16* WQT = (__bf16*)(ws + 33554432u);         //  128 KB each
  __bf16* WKT = WQT + 65536;
  __bf16* WVT = WKT + 65536;
  float*  SIG = (float*)(ws + 33554432u + 3u * 131072u);  // 64 KB

  cvt_bf16_kernel<<<4096, 256, 0, stream>>>(x, XB, BN_ * D_);
  transpose_w_kernel<<<256, 256, 0, stream>>>(Wq, WQT);
  transpose_w_kernel<<<256, 256, 0, stream>>>(Wk, WKT);
  transpose_w_kernel<<<256, 256, 0, stream>>>(Wv, WVT);
  sigma_kernel<<<2048, 256, 0, stream>>>(x, Wsigma, SIG);

  qkv_gemm_kernel<<<256, 256, 0, stream>>>(XB, WQT, QB, 0);
  qkv_gemm_kernel<<<256, 256, 0, stream>>>(XB, WKT, KB, 0);
  qkv_gemm_kernel<<<256, 256, 0, stream>>>(XB, WVT, VT, 1);

  scores_kernel<<<1024, 256, 0, stream>>>(QB, KB, S);
  softmax_batch_kernel<<<4096, 256, 0, stream>>>(S);
  z_gemm_kernel<<<256, 256, 0, stream>>>(S, VT, Z);
  prior_kernel<<<16384, 256, 0, stream>>>(eps, SIG, P);
}
